// MultiHeadSelfAttention_36412732736063
// MI455X (gfx1250) — matrix operand order repaired and checked by reading, not compiled, not run on hardware
//
#include <hip/hip_runtime.h>

#define EMBED 1024
#define HEADS 16
#define HD    64
#define BATCH 4
#define SEQ   2048

typedef __bf16 bhalf;
typedef __attribute__((ext_vector_type(16))) __bf16 v16bf;
typedef __attribute__((ext_vector_type(8)))  __bf16 v8bf;
typedef __attribute__((ext_vector_type(8)))  float  v8f;

// ---------- helpers ----------

__device__ __forceinline__ bhalf f32_to_bf16(float x) {
    return (bhalf)x;   // native v_cvt; RNE per hardware
}

__device__ __forceinline__ v8f wmma_bf16(v16bf a, v16bf b, v8f c) {
    // D = A(16x32 bf16) * B(32x16 bf16) + C(16x16 f32)
    return __builtin_amdgcn_wmma_f32_16x16x32_bf16(false, a, false, b,
                                                   (short)0, c, false, false);
}

// A-matrix 16x32 (MxK) bf16 layout: lane m = lane&15; kb = lane<16 ? 0 : 8;
// elements 0..7 -> K = kb+e, elements 8..15 -> K = kb+16+(e-8)
__device__ __forceinline__ v16bf load_A16x32(const bhalf* base, int ld, int lane) {
    int m  = lane & 15;
    int kb = (lane < 16) ? 0 : 8;
    const bhalf* p = base + (size_t)m * ld + kb;
    v8bf lo = *(const v8bf*)(p);        // 16B aligned
    v8bf hi = *(const v8bf*)(p + 16);   // 16B aligned
    return __builtin_shufflevector(lo, hi, 0, 1, 2, 3, 4, 5, 6, 7,
                                           8, 9, 10, 11, 12, 13, 14, 15);
}

__device__ __forceinline__ v16bf load_A16x32_f32(const float* base, int ld, int lane) {
    int m  = lane & 15;
    int kb = (lane < 16) ? 0 : 8;
    const float* p = base + (size_t)m * ld + kb;
    v8f lo = *(const v8f*)(p);          // 32B aligned
    v8f hi = *(const v8f*)(p + 16);     // 32B aligned
    v8bf blo = __builtin_convertvector(lo, v8bf);
    v8bf bhi = __builtin_convertvector(hi, v8bf);
    return __builtin_shufflevector(blo, bhi, 0, 1, 2, 3, 4, 5, 6, 7,
                                             8, 9, 10, 11, 12, 13, 14, 15);
}

// B-matrix 32x16 (KxN) bf16 where memory is row-major over N (src[n][k]):
// lane n = lane&15, kb = lane<16 ? 0 : 8; elements 0..7 -> K = kb+e, 8..15 -> K = kb+16+(e-8).
__device__ __forceinline__ v16bf load_B32x16(const bhalf* base, int ld, int lane) {
    int n  = lane & 15;
    int k0 = (lane < 16) ? 0 : 8;
    const bhalf* p = base + (size_t)n * ld + k0;
    v8bf lo = *(const v8bf*)(p);        // 16B aligned
    v8bf hi = *(const v8bf*)(p + 16);   // 16B aligned
    return __builtin_shufflevector(lo, hi, 0, 1, 2, 3, 4, 5, 6, 7,
                                           8, 9, 10, 11, 12, 13, 14, 15);
}

// reductions over the 16-lane half-wave groups (C/D layout: a row lives in
// lanes 0..15 or 16..31); xor masks < 16 stay inside each half.
__device__ __forceinline__ float half16_max(float v) {
    v = fmaxf(v, __shfl_xor(v, 1, 32));
    v = fmaxf(v, __shfl_xor(v, 2, 32));
    v = fmaxf(v, __shfl_xor(v, 4, 32));
    v = fmaxf(v, __shfl_xor(v, 8, 32));
    return v;
}
__device__ __forceinline__ float half16_sum(float v) {
    v += __shfl_xor(v, 1, 32);
    v += __shfl_xor(v, 2, 32);
    v += __shfl_xor(v, 4, 32);
    v += __shfl_xor(v, 8, 32);
    return v;
}

// ---------- kernel 0: fp32 -> bf16 weight conversion ----------

__global__ void cvt_bf16_kernel(const float* __restrict__ src,
                                bhalf* __restrict__ dst, int n) {
    int i = blockIdx.x * blockDim.x + threadIdx.x;
    if (i < n) dst[i] = f32_to_bf16(src[i]);
}

// ---------- kernel 1: per-head QKV projection ----------
// One wave handles one (b, h, 16-token tile): [16x64] @ [64x64]^T, 8 WMMAs per matrix.

__device__ __forceinline__ void gemm16x64(const float* x, int xld,
                                          const bhalf* W, int lane, v8f* out) {
    v16bf a0 = load_A16x32_f32(x, xld, lane);
    v16bf a1 = load_A16x32_f32(x + 32, xld, lane);
#pragma unroll
    for (int t = 0; t < 4; ++t) {
        v8f acc = {};
        acc = wmma_bf16(a0, load_B32x16(W + t * 16 * HD, HD, lane), acc);
        acc = wmma_bf16(a1, load_B32x16(W + t * 16 * HD + 32, HD, lane), acc);
        out[t] = acc;
    }
}

__global__ __launch_bounds__(256) void qkv_proj_kernel(
    const float* __restrict__ q_in, const float* __restrict__ k_in,
    const float* __restrict__ v_in,
    const bhalf* __restrict__ Wq, const bhalf* __restrict__ Wk,
    const bhalf* __restrict__ Wv,
    bhalf* __restrict__ Qb, bhalf* __restrict__ Kb, bhalf* __restrict__ Vt)
{
    int lane = threadIdx.x & 31;
    int wave = threadIdx.x >> 5;
    int tile = blockIdx.x * 8 + wave;            // 8192 tiles total
    const int NT = SEQ / 16;
    int q16 = tile % NT;
    int h   = (tile / NT) % HEADS;
    int b   = tile / (NT * HEADS);
    int q0  = q16 * 16;

    size_t in_off = ((size_t)(b * SEQ + q0)) * EMBED + (size_t)h * HD;
    size_t bh     = (size_t)b * HEADS + h;

    int ncol = lane & 15;
    int mb   = (lane < 16) ? 0 : 8;

    v8f acc[4];

    // Q: row-major bf16 [b][h][l][d]
    gemm16x64(q_in + in_off, EMBED, Wq, lane, acc);
    {
        bhalf* dst = Qb + (bh * SEQ + q0) * HD;
#pragma unroll
        for (int t = 0; t < 4; ++t)
#pragma unroll
            for (int r = 0; r < 8; ++r)
                dst[(size_t)(mb + r) * HD + t * 16 + ncol] = f32_to_bf16(acc[t][r]);
    }
    // K: row-major bf16 [b][h][l][d]
    gemm16x64(k_in + in_off, EMBED, Wk, lane, acc);
    {
        bhalf* dst = Kb + (bh * SEQ + q0) * HD;
#pragma unroll
        for (int t = 0; t < 4; ++t)
#pragma unroll
            for (int r = 0; r < 8; ++r)
                dst[(size_t)(mb + r) * HD + t * 16 + ncol] = f32_to_bf16(acc[t][r]);
    }
    // V: transposed bf16 [b][h][d][l]  (so P@V B-operand is contiguous per lane)
    gemm16x64(v_in + in_off, EMBED, Wv, lane, acc);
    {
        bhalf* dst = Vt + bh * (size_t)HD * SEQ;
#pragma unroll
        for (int t = 0; t < 4; ++t)
#pragma unroll
            for (int r = 0; r < 8; ++r)
                dst[(size_t)(t * 16 + ncol) * SEQ + q0 + mb + r] = f32_to_bf16(acc[t][r]);
    }
}

// ---------- kernel 2: flash attention ----------
// One wave per 16-query tile of one (b,h); stream keys in chunks of 32.

#define PLD 40   // padded LDS row stride (bf16 elements), 16B multiple

__global__ __launch_bounds__(256) void flash_attn_kernel(
    const bhalf* __restrict__ Qb, const bhalf* __restrict__ Kb,
    const bhalf* __restrict__ Vt, bhalf* __restrict__ Ob)
{
    __shared__ bhalf plds[8][16 * PLD];

    int lane = threadIdx.x & 31;
    int wave = threadIdx.x >> 5;
    int tile = blockIdx.x * 8 + wave;            // 8192 tiles total
    const int NT = SEQ / 16;
    int q16 = tile % NT;
    int h   = (tile / NT) % HEADS;
    int b   = tile / (NT * HEADS);
    int q0  = q16 * 16;

    size_t bh = (size_t)b * HEADS + h;
    const bhalf* Qbase = Qb + (bh * SEQ + q0) * HD;
    const bhalf* Kbase = Kb + bh * (size_t)SEQ * HD;
    const bhalf* Vbase = Vt + bh * (size_t)HD * SEQ;

    v16bf aq0 = load_A16x32(Qbase, HD, lane);
    v16bf aq1 = load_A16x32(Qbase + 32, HD, lane);

    v8f o0 = {}, o1 = {}, o2 = {}, o3 = {};
    float mrow[8], lrow[8];
#pragma unroll
    for (int r = 0; r < 8; ++r) { mrow[r] = -3.0e38f; lrow[r] = 0.0f; }

    int ncol = lane & 15;
    int mb   = (lane < 16) ? 0 : 8;
    const float scale = 0.03125f;                // 1/sqrt(EMBED)
    bhalf* pw = &plds[wave][0];

    for (int kk = 0; kk < SEQ; kk += 32) {
        // S = Q * K^T for 32 keys (two 16-key sub-tiles), f32 accumulate
        v8f s0 = {}, s1 = {};
        s0 = wmma_bf16(aq0, load_B32x16(Kbase + (size_t)kk * HD, HD, lane), s0);
        s0 = wmma_bf16(aq1, load_B32x16(Kbase + (size_t)kk * HD + 32, HD, lane), s0);
        s1 = wmma_bf16(aq0, load_B32x16(Kbase + (size_t)(kk + 16) * HD, HD, lane), s1);
        s1 = wmma_bf16(aq1, load_B32x16(Kbase + (size_t)(kk + 16) * HD + 32, HD, lane), s1);

        if (kk + 32 < SEQ) {
            __builtin_prefetch(Kbase + (size_t)(kk + 32) * HD + (lane << 3), 0, 3);
            __builtin_prefetch(Vbase + (size_t)(lane << 1) * SEQ + kk + 32, 0, 3);
        }

        // online softmax (rows live on 16-lane half-waves in C/D layout)
        float fac[8];
#pragma unroll
        for (int r = 0; r < 8; ++r) {
            float x0 = s0[r] * scale;
            float x1 = s1[r] * scale;
            float mx   = half16_max(fmaxf(x0, x1));
            float mnew = fmaxf(mrow[r], mx);
            float f  = __expf(mrow[r] - mnew);
            float p0 = __expf(x0 - mnew);
            float p1 = __expf(x1 - mnew);
            lrow[r] = lrow[r] * f + half16_sum(p0 + p1);
            mrow[r] = mnew;
            fac[r]  = f;
            int m = mb + r;   // P tile: C/D layout -> LDS row-major [16][32]
            pw[m * PLD + ncol]      = f32_to_bf16(p0);
            pw[m * PLD + 16 + ncol] = f32_to_bf16(p1);
        }
#pragma unroll
        for (int r = 0; r < 8; ++r) {
            o0[r] *= fac[r]; o1[r] *= fac[r]; o2[r] *= fac[r]; o3[r] *= fac[r];
        }

        // reload P in WMMA A 16x32 layout (wave-private LDS, in-order DS ops)
        v16bf ap = load_A16x32(pw, PLD, lane);

        // O += P * V  (V transposed: rows = d, contiguous keys per lane)
        o0 = wmma_bf16(ap, load_B32x16(Vbase + (size_t)(0 * 16) * SEQ + kk, SEQ, lane), o0);
        o1 = wmma_bf16(ap, load_B32x16(Vbase + (size_t)(1 * 16) * SEQ + kk, SEQ, lane), o1);
        o2 = wmma_bf16(ap, load_B32x16(Vbase + (size_t)(2 * 16) * SEQ + kk, SEQ, lane), o2);
        o3 = wmma_bf16(ap, load_B32x16(Vbase + (size_t)(3 * 16) * SEQ + kk, SEQ, lane), o3);
    }

    // normalize and write attention output as bf16 [b][l][h*64+d]
    float inv[8];
#pragma unroll
    for (int r = 0; r < 8; ++r) inv[r] = 1.0f / lrow[r];

    bhalf* obase = Ob + ((size_t)(b * SEQ + q0)) * EMBED + (size_t)h * HD;
#pragma unroll
    for (int r = 0; r < 8; ++r) {
        size_t row = (size_t)(mb + r) * EMBED;
        obase[row + 0 * 16 + ncol] = f32_to_bf16(o0[r] * inv[r]);
        obase[row + 1 * 16 + ncol] = f32_to_bf16(o1[r] * inv[r]);
        obase[row + 2 * 16 + ncol] = f32_to_bf16(o2[r] * inv[r]);
        obase[row + 3 * 16 + ncol] = f32_to_bf16(o3[r] * inv[r]);
    }
}

// ---------- kernel 3: output projection  out = O @ Wo^T + bo ----------
// [8192 x 1024] @ [1024 x 1024]^T; one wave per 16x64 output strip.

__global__ __launch_bounds__(256) void out_proj_kernel(
    const bhalf* __restrict__ Ob, const bhalf* __restrict__ Wo,
    const float* __restrict__ bo, float* __restrict__ out)
{
    int lane  = threadIdx.x & 31;
    int wave  = threadIdx.x >> 5;
    int mtile = blockIdx.x * 8 + wave;           // 512 m-tiles
    int n0    = blockIdx.y * 64;

    const bhalf* Abase = Ob + (size_t)mtile * 16 * EMBED;

    v8f acc[4] = {};
    for (int k0 = 0; k0 < EMBED; k0 += 32) {
        v16bf a = load_A16x32(Abase + k0, EMBED, lane);
#pragma unroll
        for (int t = 0; t < 4; ++t) {
            v16bf bm = load_B32x16(Wo + (size_t)(n0 + t * 16) * EMBED + k0, EMBED, lane);
            acc[t] = wmma_bf16(a, bm, acc[t]);
        }
    }

    int ncol = lane & 15;
    int mb   = (lane < 16) ? 0 : 8;
#pragma unroll
    for (int t = 0; t < 4; ++t) {
        float bias = bo[n0 + t * 16 + ncol];
#pragma unroll
        for (int r = 0; r < 8; ++r)
            out[((size_t)mtile * 16 + mb + r) * EMBED + n0 + t * 16 + ncol] =
                acc[t][r] + bias;
    }
}

// ---------- launch ----------

extern "C" void kernel_launch(void* const* d_in, const int* in_sizes, int n_in,
                              void* d_out, int out_size, void* d_ws, size_t ws_size,
                              hipStream_t stream) {
    (void)in_sizes; (void)n_in; (void)out_size; (void)ws_size;

    const float* values = (const float*)d_in[0];
    const float* keys   = (const float*)d_in[1];
    const float* query  = (const float*)d_in[2];
    const float* Wq     = (const float*)d_in[3];
    const float* Wk     = (const float*)d_in[4];
    const float* Wv     = (const float*)d_in[5];
    const float* Wo     = (const float*)d_in[6];
    const float* bo     = (const float*)d_in[7];
    float* out = (float*)d_out;

    // workspace layout (bf16 elements)
    bhalf* Wq_b = (bhalf*)d_ws;
    bhalf* Wk_b = Wq_b + 64 * 64;
    bhalf* Wv_b = Wk_b + 64 * 64;
    bhalf* Wo_b = Wv_b + 64 * 64;
    bhalf* Qb   = Wo_b + (size_t)EMBED * EMBED;
    bhalf* Kb   = Qb + (size_t)BATCH * HEADS * SEQ * HD;
    bhalf* Vt   = Kb + (size_t)BATCH * HEADS * SEQ * HD;
    bhalf* Ob   = Vt + (size_t)BATCH * HEADS * SEQ * HD;

    cvt_bf16_kernel<<<16, 256, 0, stream>>>(Wq, Wq_b, 64 * 64);
    cvt_bf16_kernel<<<16, 256, 0, stream>>>(Wk, Wk_b, 64 * 64);
    cvt_bf16_kernel<<<16, 256, 0, stream>>>(Wv, Wv_b, 64 * 64);
    cvt_bf16_kernel<<<4096, 256, 0, stream>>>(Wo, Wo_b, EMBED * EMBED);

    qkv_proj_kernel<<<1024, 256, 0, stream>>>(query, keys, values,
                                              Wq_b, Wk_b, Wv_b, Qb, Kb, Vt);

    flash_attn_kernel<<<1024, 256, 0, stream>>>(Qb, Kb, Vt, Ob);

    out_proj_kernel<<<dim3(64, 16), 256, 0, stream>>>(Ob, Wo_b, bo, out);
}
